// SelectSubAttention_5403068858792
// MI455X (gfx1250) — matrix operand order repaired and checked by reading, not compiled, not run on hardware
//
#include <hip/hip_runtime.h>

// ---------------- problem constants ----------------
constexpr int Bn    = 8;
constexpr int Nn    = 1024;
constexpr int HID   = 256;
constexpr int HEADS = 8;
constexpr int DH    = 32;
constexpr int EPG   = 16384;
constexpr int KTOP  = 512;

typedef __attribute__((ext_vector_type(16))) _Float16 v16h;
typedef __attribute__((ext_vector_type(8)))  _Float16 v8h;
typedef __attribute__((ext_vector_type(8)))  float    v8f;

typedef __attribute__((ext_vector_type(4))) unsigned int u32x4;
typedef __attribute__((ext_vector_type(4))) int          i32x4;
typedef __attribute__((ext_vector_type(8))) int          i32x8;

// ---------------- TDM (Tensor Data Mover) support, guarded for toolchain ----------
#if __has_builtin(__builtin_amdgcn_tensor_load_to_lds) && __has_builtin(__builtin_amdgcn_s_wait_tensorcnt)
#define ATT_USE_TDM 1
#else
#define ATT_USE_TDM 0
#endif

#if ATT_USE_TDM
// 2D f16 tile load Global -> LDS via TDM.  tile_d0 elems/row, tile_d1 rows,
// row_stride in elems.  Descriptor per CDNA5 ISA D# groups 0/1 (§8.3/8.4).
__device__ __forceinline__ void tdm_load_2d_f16(unsigned lds_byte_addr,
                                                const void* gptr,
                                                unsigned tile_d0,
                                                unsigned tile_d1,
                                                unsigned row_stride) {
  const unsigned long long ga = (unsigned long long)(size_t)gptr;
  u32x4 g0;
  g0[0] = 1u;                                              // count=1, user mode
  g0[1] = lds_byte_addr;                                   // lds_addr [63:32]
  g0[2] = (unsigned)(ga & 0xffffffffu);                    // global_addr lo
  g0[3] = (unsigned)((ga >> 32) & 0x1ffffffu) | (2u << 30);// global_addr hi | type=2
  i32x8 g1;
  g1[0] = (int)(1u << 16);          // workgroup_mask=0, data_size=1 (2 bytes)
  g1[1] = (int)(tile_d0 << 16);     // tensor_dim0[15:0] at bits 63:48
  g1[2] = (int)(tile_d1 << 16);     // tensor_dim0 hi=0, tensor_dim1[15:0] at 95:80
  g1[3] = (int)(tile_d0 << 16);     // tensor_dim1 hi=0, tile_dim0 at 127:112
  g1[4] = (int)tile_d1;             // tile_dim1 at 143:128, tile_dim2=0
  g1[5] = (int)row_stride;          // tensor_dim0_stride lo32
  g1[6] = 0;                        // stride hi, tensor_dim1_stride lo
  g1[7] = 0;
  i32x4 z4 = {0, 0, 0, 0};
#if __has_include(<hip/amd_detail/amd_gfx1250_TDM.h>)
  i32x8 z8 = {0, 0, 0, 0, 0, 0, 0, 0};
  __builtin_amdgcn_tensor_load_to_lds(g0, g1, z4, z4, z8, 0);
#else
  __builtin_amdgcn_tensor_load_to_lds(g0, g1, z4, z4, 0);
#endif
}
#endif

// xor-lane exchange via ds_swizzle (wave32, group-of-32 mode: and=0x1f, or=0, xor=XM)
template<int XM>
__device__ __forceinline__ float swz_xor(float v) {
  int i = __builtin_bit_cast(int, v);
  i = __builtin_amdgcn_ds_swizzle(i, (XM << 10) | 0x1f);
  return __builtin_bit_cast(float, i);
}
__device__ __forceinline__ float wave_sum32(float v) {
  v += swz_xor<16>(v); v += swz_xor<8>(v); v += swz_xor<4>(v);
  v += swz_xor<2>(v);  v += swz_xor<1>(v);
  return v;
}

// ---------------- prep kernels ----------------
__global__ void k_f32_to_f16(const float* __restrict__ in, _Float16* __restrict__ out, int n) {
  int i = blockIdx.x * blockDim.x + threadIdx.x;
  if (i < n) out[i] = (_Float16)in[i];
}

// Wt[n][k] = (f16) W[k][n]  (so the WMMA B-operand reads 16 contiguous k per lane)
__global__ void k_transpose_w(const float* __restrict__ W, _Float16* __restrict__ Wt) {
  int n = blockIdx.x;
  int k = threadIdx.x;
  Wt[n * HID + k] = (_Float16)W[k * HID + n];
}

// ---------------- WMMA GEMM: C[M,256] = A_f16[M,256] @ W  (Wt = W^T in f16) ----------
template<bool OUT_F16>
__global__ void k_gemm_wmma(const _Float16* __restrict__ A,
                            const _Float16* __restrict__ Wt,
                            void* __restrict__ Cout) {
  const int lane = threadIdx.x & 31;
  const int hf   = lane >> 4;        // lane half (0/1)
  const int ln   = lane & 15;
  const int tile = blockIdx.x * 4 + (threadIdx.x >> 5);
  const int mbase = (tile >> 4) * 16;
  const int nbase = (tile & 15) * 16;

  const _Float16* arow = A  + (size_t)(mbase + ln) * HID + hf * 16;  // A: row=ln, k=16*hf..+15
  const _Float16* brow = Wt + (size_t)(nbase + ln) * HID + hf * 16;  // B: col=ln, k=16*hf..+15

  v8f c = {};
#pragma unroll
  for (int kk = 0; kk < HID; kk += 32) {
    v8h alo = *(const v8h*)(arow + kk);
    v8h ahi = *(const v8h*)(arow + kk + 8);
    v16h a, b;
#pragma unroll
    for (int i = 0; i < 8; i++) { a[i] = alo[i]; a[i + 8] = ahi[i]; }
    b = *(const v16h*)(brow + kk);
    c = __builtin_amdgcn_wmma_f32_16x16x32_f16(false, a, false, b, (short)0, c, false, false);
  }
  const int col = nbase + ln;
  if (OUT_F16) {
    _Float16* C = (_Float16*)Cout;
#pragma unroll
    for (int r = 0; r < 8; r++)
      C[(size_t)(mbase + hf * 8 + r) * HID + col] = (_Float16)c[r];
  } else {
    float* C = (float*)Cout;
#pragma unroll
    for (int r = 0; r < 8; r++)
      C[(size_t)(mbase + hf * 8 + r) * HID + col] = c[r];
  }
}

// ---------------- flash attention (S^T orientation, online softmax) -----------------
// block = 128 threads (4 waves); each wave owns a 16-row q tile; grid = B*H*16
__global__ void k_attention(const _Float16* __restrict__ Qh,
                            const _Float16* __restrict__ Kh,
                            const _Float16* __restrict__ Vh,
                            const float* __restrict__ dist,
                            const unsigned char* __restrict__ mask,
                            _Float16* __restrict__ attH) {
  const int blk = blockIdx.x;
  const int qt = blk & 15;
  const int h  = (blk >> 4) & 7;
  const int b  = blk >> 7;
  const int wave = threadIdx.x >> 5;
  const int lane = threadIdx.x & 31;
  const int hf   = lane >> 4;
  const int ln   = lane & 15;
  const int q = qt * 64 + wave * 16 + ln;

  __shared__ __align__(32) _Float16 sK[64][32];    // keys chunk [key][d]
  __shared__ __align__(32) _Float16 sVt[32][72];   // V^T [d][key], padded rows

  // B operand for S^T = K·Q^T : lane column q, elements d = i + 16*hf (contiguous)
  const v16h bq = *(const v16h*)(Qh + ((size_t)(b * Nn + q)) * HID + h * DH + hf * 16);

  v8f acc0 = {}, acc1 = {};                        // O^T tiles: d=0..15 and 16..31
  const float NEG_INF = -__builtin_inff();
  float mrun = NEG_INF, lrun = 0.f;
  const float sc = 0.17677669529663689f;           // 1/sqrt(32)

  for (int kc = 0; kc < 16; kc++) {
    const int kb0 = kc * 64;

#if ATT_USE_TDM
    // Wave 0 issues the K-chunk DMA (64 rows x 32 f16, row stride HID elems);
    // all threads overlap the V^T scatter-transpose staging.
    if (threadIdx.x < 32) {
      tdm_load_2d_f16((unsigned)(size_t)&sK[0][0],
                      Kh + ((size_t)(b * Nn + kb0)) * HID + h * DH,
                      /*tile_d0=*/32, /*tile_d1=*/64, /*row_stride=*/HID);
    }
    {
      const int t = threadIdx.x;
      const int kk = t >> 1;
      const int dh = (t & 1) * 16;
      const size_t gro = ((size_t)(b * Nn + kb0 + kk)) * HID + h * DH + dh;
      v16h vv = *(const v16h*)(Vh + gro);
#pragma unroll
      for (int i = 0; i < 16; i++) sVt[dh + i][kk] = vv[i];
    }
    if (kc + 1 < 16)   // prefetch next chunk's bias rows (global_prefetch_b8)
      __builtin_prefetch(dist + ((size_t)(b * Nn + q)) * Nn + kb0 + 64 + hf * 8, 0, 1);
    if (threadIdx.x < 32) __builtin_amdgcn_s_wait_tensorcnt(0);
    __syncthreads();
#else
    { // manual cooperative stage of K and V^T into LDS (f16)
      const int t = threadIdx.x;
      const int kk = t >> 1;
      const int dh = (t & 1) * 16;
      const size_t gro = ((size_t)(b * Nn + kb0 + kk)) * HID + h * DH + dh;
      v16h kv = *(const v16h*)(Kh + gro);
      *(v16h*)&sK[kk][dh] = kv;
      v16h vv = *(const v16h*)(Vh + gro);
#pragma unroll
      for (int i = 0; i < 16; i++) sVt[dh + i][kk] = vv[i];
    }
    if (kc + 1 < 16)
      __builtin_prefetch(dist + ((size_t)(b * Nn + q)) * Nn + kb0 + 64 + hf * 8, 0, 1);
    __syncthreads();
#endif

#pragma unroll
    for (int sub = 0; sub < 2; sub++) {
      const int lk = sub * 32;
      const int kb = kb0 + lk;

      // A operands (keys as rows): two contiguous 8-half runs per lane
      v16h a0, a1;
      {
        const _Float16* r0 = &sK[lk + ln][hf * 16];
        v8h lo = *(const v8h*)r0, hi = *(const v8h*)(r0 + 8);
#pragma unroll
        for (int i = 0; i < 8; i++) { a0[i] = lo[i]; a0[i + 8] = hi[i]; }
        const _Float16* r1 = &sK[lk + 16 + ln][hf * 16];
        lo = *(const v8h*)r1; hi = *(const v8h*)(r1 + 8);
#pragma unroll
        for (int i = 0; i < 8; i++) { a1[i] = lo[i]; a1[i + 8] = hi[i]; }
      }
      v8f z = {};
      v8f s0 = __builtin_amdgcn_wmma_f32_16x16x32_f16(false, a0, false, bq, (short)0, z, false, false);
      v8f s1 = __builtin_amdgcn_wmma_f32_16x16x32_f16(false, a1, false, bq, (short)0, z, false, false);

      // bias + mask: lane holds keys kb+8*hf+r (tile0) and +16 (tile1), contiguous
      const float* dptr = dist + ((size_t)(b * Nn + q)) * Nn + kb + hf * 8;
      float4 d0a = *(const float4*)(dptr);
      float4 d0b = *(const float4*)(dptr + 4);
      float4 d1a = *(const float4*)(dptr + 16);
      float4 d1b = *(const float4*)(dptr + 20);
      const unsigned char* mp = mask + b * Nn + kb + hf * 8;
      float bias0[8] = {d0a.x, d0a.y, d0a.z, d0a.w, d0b.x, d0b.y, d0b.z, d0b.w};
      float bias1[8] = {d1a.x, d1a.y, d1a.z, d1a.w, d1b.x, d1b.y, d1b.z, d1b.w};
      float t0[8], t1[8];
      float lmax = NEG_INF;
#pragma unroll
      for (int r = 0; r < 8; r++) {
        t0[r] = mp[r]      ? (s0[r] * sc + bias0[r]) : -1e9f;
        t1[r] = mp[r + 16] ? (s1[r] * sc + bias1[r]) : -1e9f;
        lmax = fmaxf(lmax, fmaxf(t0[r], t1[r]));
      }
      lmax = fmaxf(lmax, swz_xor<16>(lmax));       // combine the two lane-halves (keys split)
      const float mnew  = fmaxf(mrun, lmax);
      const float scale = __expf(mrun - mnew);
      float p0[8], p1[8];
      float psum = 0.f;
#pragma unroll
      for (int r = 0; r < 8; r++) {
        p0[r] = __expf(t0[r] - mnew);
        p1[r] = __expf(t1[r] - mnew);
        psum += p0[r] + p1[r];
      }
      psum += swz_xor<16>(psum);
      lrun = lrun * scale + psum;
      mrun = mnew;
#pragma unroll
      for (int r = 0; r < 8; r++) { acc0[r] *= scale; acc1[r] *= scale; }

      // Build B operand P^T[32 keys x 16 q] from accumulator layout via xor-16 exchange
      v16h bp;
#pragma unroll
      for (int r = 0; r < 8; r++) {
        const float o0 = swz_xor<16>(p0[r]);
        const float o1 = swz_xor<16>(p1[r]);
        bp[r]     = (_Float16)((hf == 0) ? p0[r] : o1);
        bp[r + 8] = (_Float16)((hf == 0) ? o0    : p1[r]);
      }

      // A operands from V^T (rows = d), contiguous key runs
      v16h av0, av1;
      {
        const _Float16* v0p = &sVt[ln][lk + hf * 16];
        v8h lo = *(const v8h*)v0p, hi = *(const v8h*)(v0p + 8);
#pragma unroll
        for (int i = 0; i < 8; i++) { av0[i] = lo[i]; av0[i + 8] = hi[i]; }
        const _Float16* v1p = &sVt[16 + ln][lk + hf * 16];
        lo = *(const v8h*)v1p; hi = *(const v8h*)(v1p + 8);
#pragma unroll
        for (int i = 0; i < 8; i++) { av1[i] = lo[i]; av1[i + 8] = hi[i]; }
      }
      acc0 = __builtin_amdgcn_wmma_f32_16x16x32_f16(false, av0, false, bp, (short)0, acc0, false, false);
      acc1 = __builtin_amdgcn_wmma_f32_16x16x32_f16(false, av1, false, bp, (short)0, acc1, false, false);
    }
    __syncthreads();
  }

  const float inv = (lrun > 0.f) ? (1.f / lrun) : 0.f;
  v8h o0, o1;
#pragma unroll
  for (int r = 0; r < 8; r++) {
    o0[r] = (_Float16)(acc0[r] * inv);
    o1[r] = (_Float16)(acc1[r] * inv);
  }
  _Float16* op = attH + ((size_t)(b * Nn + q)) * HID + h * DH;
  *(v8h*)(op + hf * 8) = o0;           // d = 8*hf + r
  *(v8h*)(op + 16 + hf * 8) = o1;      // d = 16 + 8*hf + r
}

// ---------------- TopK pooling -----------------------------------------------------
__global__ void k_node_score(const float* __restrict__ enc,
                             const float* __restrict__ w,
                             float* __restrict__ s) {
  const int lane = threadIdx.x & 31;
  const int node = blockIdx.x * 4 + (threadIdx.x >> 5);
  const float* x = enc + (size_t)node * HID;
  float dot = 0.f, wsq = 0.f;
#pragma unroll
  for (int i = lane; i < HID; i += 32) {
    const float wi = w[i];
    dot += x[i] * wi;
    wsq += wi * wi;
  }
  dot = wave_sum32(dot);
  wsq = wave_sum32(wsq);
  if (lane == 0) s[node] = tanhf(dot * rsqrtf(wsq));
}

// per-graph bitonic sort of 1024 (score,idx), desc with index tiebreak (stable top_k)
__global__ void k_topk_sort(const float* __restrict__ s,
                            long long* __restrict__ out_perm,
                            float* __restrict__ svals,
                            int* __restrict__ node_map) {
  const int b = blockIdx.x;
  const int tid = threadIdx.x;
  __shared__ float v[Nn];
  __shared__ int  ix[Nn];
  v[tid] = s[b * Nn + tid];
  ix[tid] = tid;
  __syncthreads();
  for (int k = 2; k <= Nn; k <<= 1) {
    for (int j = k >> 1; j > 0; j >>= 1) {
      const int p = tid ^ j;
      if (p > tid) {
        const float va = v[tid], vb = v[p];
        const int ia = ix[tid], ib = ix[p];
        const bool aPre = (va > vb) || (va == vb && ia < ib);
        const bool desired = ((tid & k) == 0);
        if (desired ? !aPre : aPre) {
          v[tid] = vb; v[p] = va; ix[tid] = ib; ix[p] = ia;
        }
      }
      __syncthreads();
    }
  }
  node_map[b * Nn + tid] = -1;
  __syncthreads();
  if (tid < KTOP) {
    const int nd = ix[tid];
    out_perm[b * KTOP + tid] = (long long)nd;
    svals[b * KTOP + tid] = v[tid];
    node_map[b * Nn + nd] = tid;
  }
}

__global__ void k_gather_subx(const float* __restrict__ enc,
                              const long long* __restrict__ perm,
                              const float* __restrict__ svals,
                              float* __restrict__ subx) {
  const int row = blockIdx.x;            // b*KTOP + i
  const int c = threadIdx.x;
  const int b = row >> 9;
  const int nd = (int)perm[row];
  subx[(size_t)row * HID + c] = enc[((size_t)b * Nn + nd) * HID + c] * svals[row];
}

// per-graph: remap edges, bitonic sort 16384 packed keys in 128KB dynamic LDS
__global__ void k_edge_sort(const long long* __restrict__ edges,
                            const int* __restrict__ node_map,
                            long long* __restrict__ out_e,
                            unsigned char* __restrict__ out_valid) {
  extern __shared__ unsigned long long key[];   // EPG entries
  __shared__ int snm[Nn];
  const int b = blockIdx.x;
  const int tid = threadIdx.x;
  snm[tid] = node_map[b * Nn + tid];
  __syncthreads();
  const long long off = (long long)b * Nn;
#pragma unroll
  for (int sblk = 0; sblk < EPG / 1024; sblk++) {
    const int e = tid + sblk * 1024;
    const int ge = b * EPG + e;
    const int u = (int)(edges[ge] - off);
    const int w = (int)(edges[(size_t)Bn * EPG + ge] - off);
    const int nu = snm[u], nv = snm[w];
    const bool valid = (nu >= 0) && (nv >= 0);
    const unsigned int sk = valid ? (unsigned)(nu * KTOP + nv) : (unsigned)(KTOP * KTOP);
    key[e] = ((unsigned long long)sk << 14) | (unsigned)e;   // stable: tiebreak on edge id
  }
  __syncthreads();
  for (int k = 2; k <= EPG; k <<= 1) {
    for (int j = k >> 1; j > 0; j >>= 1) {
      for (int sblk = 0; sblk < EPG / 1024; sblk++) {
        const int i = tid + sblk * 1024;
        const int p = i ^ j;
        if (p > i) {
          const unsigned long long a = key[i];
          const unsigned long long c = key[p];
          const bool asc = ((i & k) == 0);
          if (asc ? (a > c) : (a < c)) { key[i] = c; key[p] = a; }
        }
      }
      __syncthreads();
    }
  }
#pragma unroll
  for (int sblk = 0; sblk < EPG / 1024; sblk++) {
    const int e = tid + sblk * 1024;
    const unsigned long long kv = key[e];
    const unsigned int sk = (unsigned int)(kv >> 14);
    const bool valid = sk != (unsigned)(KTOP * KTOP);
    out_e[((size_t)b * 2 + 0) * EPG + e] = valid ? (long long)(sk / KTOP) : -1ll;
    out_e[((size_t)b * 2 + 1) * EPG + e] = valid ? (long long)(sk % KTOP) : -1ll;
    out_valid[(size_t)b * EPG + e] = valid ? 1 : 0;
  }
}

// ---------------- launcher ---------------------------------------------------------
extern "C" void kernel_launch(void* const* d_in, const int* in_sizes, int n_in,
                              void* d_out, int out_size, void* d_ws, size_t ws_size,
                              hipStream_t stream) {
  (void)in_sizes; (void)n_in; (void)out_size; (void)ws_size;

  const float*         emb        = (const float*)d_in[0];
  const long long*     edge_index = (const long long*)d_in[1];
  const unsigned char* mask       = (const unsigned char*)d_in[2];
  const float*         dist       = (const float*)d_in[3];
  const float*         Wq         = (const float*)d_in[5];
  const float*         Wk         = (const float*)d_in[6];
  const float*         Wv         = (const float*)d_in[7];
  const float*         Wo         = (const float*)d_in[8];
  const float*         topk_w     = (const float*)d_in[9];

  const int M = Bn * Nn;   // 8192 rows

  // outputs, byte-packed in return order with native dtypes
  float*         out_enc   = (float*)d_out;
  float*         out_subx  = out_enc + (size_t)M * HID;
  long long*     out_edges = (long long*)(out_subx + (size_t)Bn * KTOP * HID);
  long long*     out_perm  = out_edges + (size_t)Bn * 2 * EPG;
  unsigned char* out_valid = (unsigned char*)(out_perm + (size_t)Bn * KTOP);

  // workspace carve-out (~21 MB)
  char* p = (char*)d_ws;
  auto take = [&](size_t bytes) { char* r = p; p += (bytes + 255) & ~(size_t)255; return r; };
  _Float16* Xh   = (_Float16*)take((size_t)M * HID * 2);
  _Float16* Wqt  = (_Float16*)take((size_t)HID * HID * 2);
  _Float16* Wkt  = (_Float16*)take((size_t)HID * HID * 2);
  _Float16* Wvt  = (_Float16*)take((size_t)HID * HID * 2);
  _Float16* Wot  = (_Float16*)take((size_t)HID * HID * 2);
  _Float16* Qh   = (_Float16*)take((size_t)M * HID * 2);
  _Float16* Kh   = (_Float16*)take((size_t)M * HID * 2);
  _Float16* Vh   = (_Float16*)take((size_t)M * HID * 2);
  _Float16* AttH = (_Float16*)take((size_t)M * HID * 2);
  float*    scores   = (float*)take((size_t)M * 4);
  float*    svals    = (float*)take((size_t)Bn * KTOP * 4);
  int*      node_map = (int*)take((size_t)M * 4);

  // 1. precision staging
  k_f32_to_f16<<<(M * HID + 255) / 256, 256, 0, stream>>>(emb, Xh, M * HID);
  k_transpose_w<<<HID, HID, 0, stream>>>(Wq, Wqt);
  k_transpose_w<<<HID, HID, 0, stream>>>(Wk, Wkt);
  k_transpose_w<<<HID, HID, 0, stream>>>(Wv, Wvt);
  k_transpose_w<<<HID, HID, 0, stream>>>(Wo, Wot);

  // 2. QKV projections (WMMA, f16 out)
  const int gemmBlocks = (M / 16) * (HID / 16) / 4;   // 2048
  k_gemm_wmma<true><<<gemmBlocks, 128, 0, stream>>>(Xh, Wqt, Qh);
  k_gemm_wmma<true><<<gemmBlocks, 128, 0, stream>>>(Xh, Wkt, Kh);
  k_gemm_wmma<true><<<gemmBlocks, 128, 0, stream>>>(Xh, Wvt, Vh);

  // 3. fused flash attention (WMMA + TDM staging + online softmax), then out proj
  k_attention<<<Bn * HEADS * 16, 128, 0, stream>>>(Qh, Kh, Vh, dist, mask, AttH);
  k_gemm_wmma<false><<<gemmBlocks, 128, 0, stream>>>(AttH, Wot, out_enc);

  // 4. TopK pooling
  k_node_score<<<M / 4, 128, 0, stream>>>(out_enc, topk_w, scores);
  k_topk_sort<<<Bn, Nn, 0, stream>>>(scores, out_perm, svals, node_map);
  k_gather_subx<<<Bn * KTOP, HID, 0, stream>>>(out_enc, out_perm, svals, out_subx);
  k_edge_sort<<<Bn, 1024, EPG * sizeof(unsigned long long), stream>>>(
      edge_index, node_map, out_edges, out_valid);
}
